// CausalDit_30571577213496
// MI455X (gfx1250) — matrix operand order repaired and checked by reading, compiled, not run on hardware
//
#include <hip/hip_runtime.h>
#include <hip/hip_bf16.h>
#include <cstddef>

// ---------------- model constants ----------------
#define DM      512
#define NH      8
#define DH      64
#define SEQ     257          // 256 patch tokens + 1 (register or action)
#define ZR_ROWS 1542         // 6*257
#define XA_ROWS 1285         // 5*257
#define KV_ROWS 2827         // ZR_ROWS + XA_ROWS
#define HIDF    2048
#define NBLK    6
#define SPAD_C  2832         // padded score stride (cross), mult of 16 halfs
#define SPAD_S  1296         // padded score stride (self)

typedef __attribute__((ext_vector_type(16))) _Float16 v16h;
typedef __attribute__((ext_vector_type(8)))  _Float16 v8h;
typedef __attribute__((ext_vector_type(4)))  _Float16 v4h;
typedef __attribute__((ext_vector_type(8)))  float    v8f;

// =====================================================================
// f32 -> f16 bulk conversion (4 elements/thread, b128 load + b64 store)
// =====================================================================
__global__ void f32_to_f16_kernel(_Float16* __restrict__ dst,
                                  const float* __restrict__ src, int n4) {
  int i = blockIdx.x * blockDim.x + threadIdx.x;
  if (i >= n4) return;
  float4 v = ((const float4*)src)[i];
  v4h h;
  h[0] = (_Float16)v.x; h[1] = (_Float16)v.y;
  h[2] = (_Float16)v.z; h[3] = (_Float16)v.w;
  ((v4h*)dst)[i] = h;
}

// =====================================================================
// Patch embed: build zr (zt + pe, register token) and xa (xt + pe, action)
// =====================================================================
__global__ void embed_kernel(const float* __restrict__ z,
                             const float* __restrict__ frames,
                             const int*   __restrict__ actions,
                             const float* __restrict__ Wp,
                             const float* __restrict__ bp,
                             const float* __restrict__ regs,
                             const float* __restrict__ pe,
                             const float* __restrict__ act_emb,
                             float* __restrict__ zr,
                             float* __restrict__ xa) {
  int row = blockIdx.x;
  int tid = threadIdx.x;
  __shared__ float p[12];
  bool is_z = row < ZR_ROWS;
  int r2 = is_z ? row : (row - ZR_ROWS);
  int f = r2 / SEQ, s = r2 % SEQ;
  float* dst = (is_z ? zr : xa) + (size_t)r2 * DM;
  if (s < 256 && tid < 12) {
    int c = tid >> 2, py = (tid >> 1) & 1, px = tid & 1;
    int hp = s >> 4, wp = s & 15;
    int hh = hp * 2 + py, ww = wp * 2 + px;
    const float* img = is_z ? z : frames;
    p[tid] = img[((size_t)(f * 3 + c) * 32 + hh) * 32 + ww];
  }
  __syncthreads();
  if (s < 256) {
    for (int d = tid; d < DM; d += blockDim.x) {
      float acc = bp[d] + pe[(size_t)s * DM + d];
      #pragma unroll
      for (int k = 0; k < 12; ++k) acc += p[k] * Wp[k * DM + d];
      dst[d] = acc;
    }
  } else if (is_z) {
    for (int d = tid; d < DM; d += blockDim.x) dst[d] = regs[d];
  } else {
    int a = actions[f];
    for (int d = tid; d < DM; d += blockDim.x) dst[d] = act_emb[(size_t)a * DM + d];
  }
}

// =====================================================================
// cond = time_emb[ts], clean = time_emb[0], plus silu() versions
// =====================================================================
__global__ void build_cond_kernel(const int* __restrict__ ts,
                                  const float* __restrict__ time_emb,
                                  float* cond, float* sil_cond,
                                  float* clean, float* sil_clean) {
  int o = blockIdx.x * blockDim.x + threadIdx.x;
  if (o < 6 * DM) {
    int f = o / DM, d = o % DM;
    float v = time_emb[(size_t)ts[f] * DM + d];
    cond[o] = v;
    sil_cond[o] = v / (1.f + __expf(-v));
  }
  int o2 = o - 6 * DM;
  if (o2 >= 0 && o2 < 5 * DM) {
    int d = o2 % DM;
    float v = time_emb[d];
    clean[o2] = v;
    sil_clean[o2] = v / (1.f + __expf(-v));
  }
}

// =====================================================================
// tiny GEMM (M<=6)
// =====================================================================
__global__ void small_mm_kernel(const float* __restrict__ A,
                                const float* __restrict__ W,
                                const float* __restrict__ bias,
                                float* __restrict__ out,
                                int M, int N, int K) {
  int idx = blockIdx.x * blockDim.x + threadIdx.x;
  if (idx >= M * N) return;
  int m = idx / N, n = idx % N;
  float acc = bias[n];
  for (int k = 0; k < K; ++k) acc += A[(size_t)m * K + k] * W[(size_t)k * N + n];
  out[idx] = acc;
}

// =====================================================================
// LayerNorm + AdaLN modulation; writes f32 and f16 copies
// =====================================================================
__global__ void ln_mod_kernel(float* __restrict__ out,
                              _Float16* __restrict__ outh,
                              const float* __restrict__ in,
                              const float* __restrict__ modvec,
                              int seq) {
  int row = blockIdx.x, tid = threadIdx.x;
  const float* x = in + (size_t)row * DM;
  __shared__ float red[256];
  float v0 = x[tid], v1 = x[tid + 256];
  red[tid] = v0 + v1;
  __syncthreads();
  for (int s2 = 128; s2 > 0; s2 >>= 1) { if (tid < s2) red[tid] += red[tid + s2]; __syncthreads(); }
  float mu = red[0] * (1.f / DM);
  __syncthreads();
  float c0 = v0 - mu, c1 = v1 - mu;
  red[tid] = c0 * c0 + c1 * c1;
  __syncthreads();
  for (int s2 = 128; s2 > 0; s2 >>= 1) { if (tid < s2) red[tid] += red[tid + s2]; __syncthreads(); }
  float rstd = rsqrtf(red[0] * (1.f / DM) + 1e-5f);
  int f = row / seq;
  const float* mv = modvec + (size_t)f * 2 * DM;
  float r0 = c0 * rstd * (1.f + mv[tid])       + mv[DM + tid];
  float r1 = c1 * rstd * (1.f + mv[tid + 256]) + mv[DM + tid + 256];
  out[(size_t)row * DM + tid]        = r0;
  out[(size_t)row * DM + tid + 256]  = r1;
  outh[(size_t)row * DM + tid]       = (_Float16)r0;
  outh[(size_t)row * DM + tid + 256] = (_Float16)r1;
}

// =====================================================================
// out = x * gvec[frame] (+ add)     (all f32)
// =====================================================================
__global__ void gate_add_kernel(float* __restrict__ out,
                                const float* __restrict__ x,
                                const float* __restrict__ g,
                                const float* __restrict__ add,
                                int seq) {
  int row = blockIdx.x, tid = threadIdx.x;
  int f = row / seq;
  for (int d = tid; d < DM; d += 256) {
    float v = x[(size_t)row * DM + d] * g[(size_t)f * DM + d];
    if (add) v += add[(size_t)row * DM + d];
    out[(size_t)row * DM + d] = v;
  }
}

// =====================================================================
// GEGLU activation: reads f32 hidden, writes f16 for the next GEMM
// =====================================================================
__global__ void geglu_act_kernel(_Float16* __restrict__ out,
                                 const float* __restrict__ h) {
  int row = blockIdx.x, tid = threadIdx.x;
  for (int j = tid; j < HIDF; j += 256) {
    float a = h[(size_t)row * (2 * HIDF) + j];
    float g = h[(size_t)row * (2 * HIDF) + HIDF + j];
    float ge = 0.5f * g * (1.f + erff(g * 0.70710678118654752f));
    out[(size_t)row * HIDF + j] = (_Float16)(a * ge);
  }
}

// =====================================================================
// masked softmax over f32 logits, writes normalized f16 probabilities
// mode 0: cross, mode 1: self
// =====================================================================
__global__ void softmax_mask_kernel(float* __restrict__ S,
                                    _Float16* __restrict__ P,
                                    int lk, int ld, int mode) {
  int q = blockIdx.x, tid = threadIdx.x;
  float* row = S + (size_t)q * ld;
  _Float16* prow = P + (size_t)q * ld;
  int qf = q / SEQ;
  __shared__ float red[256];
  float mx = -3.4e38f;
  for (int k = tid; k < lk; k += 256) {
    bool allowed;
    if (mode == 0) {
      if (k < ZR_ROWS) {
        allowed = (k / SEQ) == qf;                 // m_left: same frame
      } else {
        int j = (k - ZR_ROWS) / SEQ;               // m_right: window of 4 prev frames
        allowed = (j >= qf - 4) && (j < qf);
      }
    } else {
      allowed = (k / SEQ) <= qf;                   // causal over frames
    }
    float v = allowed ? row[k] : -1e9f;
    row[k] = v;
    mx = fmaxf(mx, v);
  }
  red[tid] = mx; __syncthreads();
  for (int s2 = 128; s2 > 0; s2 >>= 1) { if (tid < s2) red[tid] = fmaxf(red[tid], red[tid + s2]); __syncthreads(); }
  mx = red[0]; __syncthreads();
  float sum = 0.f;
  for (int k = tid; k < lk; k += 256) { float e = __expf(row[k] - mx); row[k] = e; sum += e; }
  red[tid] = sum; __syncthreads();
  for (int s2 = 128; s2 > 0; s2 >>= 1) { if (tid < s2) red[tid] += red[tid + s2]; __syncthreads(); }
  float inv = 1.f / red[0];
  for (int k = tid; k < lk; k += 256) prow[k] = (_Float16)(row[k] * inv);
}

// =====================================================================
// WMMA GEMM, f16 in / f32 accumulate. Tile 256(M)x128(N), Kstep 32.
// 8 wave32 waves in a 4x2 grid; each wave computes 64x64 = 4x4 WMMA
// tiles (16 v_wmma per K-step; 32 FLOP per LDS byte).
// Fills use b128 global loads + b128 LDS stores; prefetch next K tile.
// bT=1 -> B stored row-major as [N,K] (Q*K^T). C (f32) / Ch (f16) optional.
// =====================================================================
#define GT_M 256
#define GT_N 128
#define GT_K 32
#define LSTR 40   // LDS stride in halfs (row byte stride 80 = 16B aligned)

__global__ __launch_bounds__(256)
void gemm_kernel(const _Float16* __restrict__ A, const _Float16* __restrict__ B,
                 float* __restrict__ C, _Float16* __restrict__ Ch,
                 const float* __restrict__ bias,
                 int M, int N, int K, int lda, int ldb, int ldc,
                 float alpha, int bT) {
  __shared__ _Float16 As[GT_M * LSTR];   // 20 KB
  __shared__ _Float16 Bs[GT_N * LSTR];   // 10 KB
  int tid = threadIdx.x;
  int lane = tid & 31;
  int wave = tid >> 5;
  int m0 = blockIdx.y * GT_M;
  int n0 = blockIdx.x * GT_N;
  int mw = (wave >> 1) * 64;   // 4 wave-rows of 64
  int nw = (wave & 1) * 64;    // 2 wave-cols of 64
  int hs = lane >> 4;
  int lr = lane & 15;

  v8f acc[4][4];
  #pragma unroll
  for (int i = 0; i < 4; ++i)
    #pragma unroll
    for (int j = 0; j < 4; ++j)
      #pragma unroll
      for (int e = 0; e < 8; ++e) acc[i][j][e] = 0.f;

  for (int k0 = 0; k0 < K; k0 += GT_K) {
    // speculative prefetch of next K tile (global_prefetch_b8)
    int k1 = k0 + GT_K;
    if (k1 < K) {
      int pm = m0 + tid;
      if (pm < M) __builtin_prefetch(A + (size_t)pm * lda + k1, 0, 1);
      if (bT) {
        int pn = n0 + (tid >> 1);
        if (pn < N) __builtin_prefetch(B + (size_t)pn * ldb + k1 + (tid & 1) * 16, 0, 1);
      } else {
        int pk = k1 + (tid & 31);
        int pn = n0 + ((tid >> 5) & 7) * 16;
        if (pk < K && pn < N) __builtin_prefetch(B + (size_t)pk * ldb + pn, 0, 1);
      }
    }
    __syncthreads();
    // ---- A tile fill: 256x32, 8 halfs per thread-iteration (b128) ----
    for (int i = tid; i < (GT_M * GT_K) / 8; i += 256) {
      int m  = i >> 2;
      int kq = (i & 3) * 8;
      int gm = m0 + m, gk = k0 + kq;
      v8h v;
      if (gm < M && gk + 7 < K) {
        v = *(const v8h*)(A + (size_t)gm * lda + gk);
      } else {
        #pragma unroll
        for (int e = 0; e < 8; ++e)
          v[e] = (gm < M && gk + e < K) ? A[(size_t)gm * lda + gk + e] : (_Float16)0.f;
      }
      *(v8h*)&As[m * LSTR + kq] = v;
    }
    // ---- B tile fill into Bs[n][k] ----
    if (bT) {
      for (int i = tid; i < (GT_N * GT_K) / 8; i += 256) {
        int n  = i >> 2;
        int kq = (i & 3) * 8;
        int gn = n0 + n, gk = k0 + kq;
        v8h v;
        if (gn < N && gk + 7 < K) {
          v = *(const v8h*)(B + (size_t)gn * ldb + gk);
        } else {
          #pragma unroll
          for (int e = 0; e < 8; ++e)
            v[e] = (gn < N && gk + e < K) ? B[(size_t)gn * ldb + gk + e] : (_Float16)0.f;
        }
        *(v8h*)&Bs[n * LSTR + kq] = v;
      }
    } else {
      for (int i = tid; i < (GT_K * GT_N) / 8; i += 256) {
        int k  = i >> 4;
        int nq = (i & 15) * 8;
        int gk = k0 + k, gn = n0 + nq;
        v8h v;
        if (gk < K && gn + 7 < N) {
          v = *(const v8h*)(B + (size_t)gk * ldb + gn);
        } else {
          #pragma unroll
          for (int e = 0; e < 8; ++e)
            v[e] = (gk < K && gn + e < N) ? B[(size_t)gk * ldb + gn + e] : (_Float16)0.f;
        }
        #pragma unroll
        for (int e = 0; e < 8; ++e) Bs[(nq + e) * LSTR + k] = v[e];
      }
    }
    __syncthreads();

    // ---- fragments (contiguous k = 16*hs + s, same order for A and B), compiler merges into ds_load_b128
    v16h af[4], bf[4];
    #pragma unroll
    for (int t = 0; t < 4; ++t) {
      const _Float16* base = &As[(mw + 16 * t + lr) * LSTR];
      #pragma unroll
      for (int j = 0; j < 8; ++j) {
        int kk = 16 * hs + 2 * j;
        af[t][2 * j]     = base[kk];
        af[t][2 * j + 1] = base[kk + 1];
      }
    }
    #pragma unroll
    for (int t = 0; t < 4; ++t) {
      const _Float16* base = &Bs[(nw + 16 * t + lr) * LSTR];
      #pragma unroll
      for (int j = 0; j < 8; ++j) {
        int kk = 16 * hs + 2 * j;
        bf[t][2 * j]     = base[kk];
        bf[t][2 * j + 1] = base[kk + 1];
      }
    }
    #pragma unroll
    for (int i = 0; i < 4; ++i)
      #pragma unroll
      for (int j = 0; j < 4; ++j)
        acc[i][j] = __builtin_amdgcn_wmma_f32_16x16x32_f16(
            false, af[i], false, bf[j], (short)0, acc[i][j], false, false);
  }

  // ---- epilogue: VGPR r -> row r + 8*hs, lane lr -> col ----
  #pragma unroll
  for (int i = 0; i < 4; ++i)
    #pragma unroll
    for (int j = 0; j < 4; ++j)
      #pragma unroll
      for (int r = 0; r < 8; ++r) {
        int gm = m0 + mw + 16 * i + r + 8 * hs;
        int gn = n0 + nw + 16 * j + lr;
        if (gm < M && gn < N) {
          float v = alpha * acc[i][j][r];
          if (bias) v += bias[gn];
          if (C)  C[(size_t)gm * ldc + gn] = v;
          if (Ch) Ch[(size_t)gm * ldc + gn] = (_Float16)v;
        }
      }
}

// =====================================================================
// unpatch
// =====================================================================
__global__ void unpatch_kernel(const float* __restrict__ zr,
                               const float* __restrict__ Wu,
                               const float* __restrict__ bu,
                               float* __restrict__ out) {
  int o = blockIdx.x * blockDim.x + threadIdx.x;
  if (o >= 6 * 3 * 32 * 32) return;
  int f = o / (3 * 32 * 32);
  int rem = o % (3 * 32 * 32);
  int c = rem / (32 * 32);
  int hh = (rem % (32 * 32)) / 32;
  int ww = rem % 32;
  int s = (hh >> 1) * 16 + (ww >> 1);
  int idx = c * 4 + (hh & 1) * 2 + (ww & 1);
  const float* xr = zr + (size_t)(f * SEQ + s) * DM;
  float acc = bu[idx];
  for (int d = 0; d < DM; ++d) acc += xr[d] * Wu[(size_t)d * 12 + idx];
  out[o] = acc;
}

// ---------------------------------------------------------------------
static void gemm(hipStream_t st, const _Float16* A, const _Float16* B,
                 float* C, _Float16* Ch, const float* bias,
                 int M, int N, int K, int lda, int ldb, int ldc,
                 float alpha, int bT) {
  dim3 grid((N + GT_N - 1) / GT_N, (M + GT_M - 1) / GT_M);
  gemm_kernel<<<grid, 256, 0, st>>>(A, B, C, Ch, bias, M, N, K, lda, ldb, ldc, alpha, bT);
}

static void small_mm(hipStream_t st, const float* A, const float* W,
                     const float* b, float* out, int M, int N, int K) {
  small_mm_kernel<<<(M * N + 255) / 256, 256, 0, st>>>(A, W, b, out, M, N, K);
}

static void to_half(hipStream_t st, _Float16* dst, const float* src, size_t n) {
  int n4 = (int)(n / 4);
  f32_to_f16_kernel<<<(n4 + 255) / 256, 256, 0, st>>>(dst, src, n4);
}

extern "C" void kernel_launch(void* const* d_in, const int* in_sizes, int n_in,
                              void* d_out, int out_size, void* d_ws, size_t ws_size,
                              hipStream_t stream) {
  const float* z         = (const float*)d_in[0];
  const float* frames    = (const float*)d_in[1];
  const int*   actions   = (const int*)  d_in[2];
  const int*   ts        = (const int*)  d_in[3];
  const float* W_patch   = (const float*)d_in[4];
  const float* b_patch   = (const float*)d_in[5];
  const float* W_unpatch = (const float*)d_in[6];
  const float* b_unpatch = (const float*)d_in[7];
  const float* registers = (const float*)d_in[8];
  const float* pe_grid   = (const float*)d_in[9];
  const float* action_em = (const float*)d_in[10];
  const float* time_emb  = (const float*)d_in[11];
  const float* W_mod1    = (const float*)d_in[12];
  const float* b_mod1    = (const float*)d_in[13];
  const float* W_mod2    = (const float*)d_in[14];
  const float* b_mod2    = (const float*)d_in[15];
  const float* W_q       = (const float*)d_in[16];
  const float* b_q       = (const float*)d_in[17];
  const float* W_k       = (const float*)d_in[18];
  const float* b_k       = (const float*)d_in[19];
  const float* W_v       = (const float*)d_in[20];
  const float* b_v       = (const float*)d_in[21];
  const float* W_o       = (const float*)d_in[22];
  const float* b_o       = (const float*)d_in[23];
  const float* W_g1      = (const float*)d_in[24];
  const float* b_g1      = (const float*)d_in[25];
  const float* W_g2      = (const float*)d_in[26];
  const float* b_g2      = (const float*)d_in[27];
  const float* W_geglu   = (const float*)d_in[28];
  const float* b_geglu   = (const float*)d_in[29];
  const float* W_ffout   = (const float*)d_in[30];
  const float* b_ffout   = (const float*)d_in[31];

  // ---------------- workspace carve ----------------
  float* w = (float*)d_ws;
  size_t off = 0;
  auto alloc = [&](size_t n) { float* p = w + off; off += n; return p; };
  float* zr    = alloc((size_t)ZR_ROWS * DM);
  float* xa    = alloc((size_t)XA_ROWS * DM);
  float* zrn   = alloc((size_t)ZR_ROWS * DM);
  float* xan   = alloc((size_t)XA_ROWS * DM);
  float* tmpz  = alloc((size_t)ZR_ROWS * DM);
  float* tmpx  = alloc((size_t)XA_ROWS * DM);
  float* scores= alloc((size_t)ZR_ROWS * SPAD_C);
  float* hid   = alloc((size_t)ZR_ROWS * 2 * HIDF);
  float* cond  = alloc(6 * DM);
  float* silc  = alloc(6 * DM);
  float* clean = alloc(5 * DM);
  float* silx  = alloc(5 * DM);
  float* mv1c  = alloc(6 * 2 * DM);
  float* mv1x  = alloc(5 * 2 * DM);
  float* mv2c  = alloc(6 * 2 * DM);
  float* mv2x  = alloc(5 * 2 * DM);
  float* g1c   = alloc(6 * DM);
  float* g1x   = alloc(5 * DM);
  float* g2c   = alloc(6 * DM);
  float* g2x   = alloc(5 * DM);

  _Float16* hb = (_Float16*)(w + off);
  size_t hoff = 0;
  auto halloc = [&](size_t n) { _Float16* p = hb + hoff; hoff += n; return p; };
  _Float16* zrn_h  = halloc((size_t)ZR_ROWS * DM);
  _Float16* xan_h  = halloc((size_t)XA_ROWS * DM);
  _Float16* qz_h   = halloc((size_t)ZR_ROWS * DM);
  _Float16* qx_h   = halloc((size_t)XA_ROWS * DM);
  _Float16* kall_h = halloc((size_t)KV_ROWS * DM);
  _Float16* vall_h = halloc((size_t)KV_ROWS * DM);
  _Float16* attnz_h= halloc((size_t)ZR_ROWS * DM);
  _Float16* attnx_h= halloc((size_t)XA_ROWS * DM);
  _Float16* prob_h = halloc((size_t)ZR_ROWS * SPAD_C);
  _Float16* hact_h = halloc((size_t)ZR_ROWS * HIDF);
  _Float16* Wq_h   = halloc((size_t)NBLK * DM * DM);
  _Float16* Wk_h   = halloc((size_t)NBLK * DM * DM);
  _Float16* Wv_h   = halloc((size_t)NBLK * DM * DM);
  _Float16* Wo_h   = halloc((size_t)NBLK * DM * DM);
  _Float16* Wgg_h  = halloc((size_t)NBLK * DM * 2 * HIDF);
  _Float16* Wff_h  = halloc((size_t)NBLK * HIDF * DM);
  (void)ws_size; (void)n_in; (void)in_sizes; (void)out_size;

  // ---------------- one-time weight conversion ----------------
  to_half(stream, Wq_h,  W_q,     (size_t)NBLK * DM * DM);
  to_half(stream, Wk_h,  W_k,     (size_t)NBLK * DM * DM);
  to_half(stream, Wv_h,  W_v,     (size_t)NBLK * DM * DM);
  to_half(stream, Wo_h,  W_o,     (size_t)NBLK * DM * DM);
  to_half(stream, Wgg_h, W_geglu, (size_t)NBLK * DM * 2 * HIDF);
  to_half(stream, Wff_h, W_ffout, (size_t)NBLK * HIDF * DM);

  // ---------------- embeddings + cond ----------------
  embed_kernel<<<ZR_ROWS + XA_ROWS, 256, 0, stream>>>(
      z, frames, actions, W_patch, b_patch, registers, pe_grid, action_em, zr, xa);
  build_cond_kernel<<<(11 * DM + 255) / 256, 256, 0, stream>>>(
      ts, time_emb, cond, silc, clean, silx);

  for (int i = 0; i < NBLK; ++i) {
    const float* bm1 = b_mod1 + (size_t)i * 2 * DM;
    const float* bm2 = b_mod2 + (size_t)i * 2 * DM;
    const float* Wm1 = W_mod1 + (size_t)i * DM * 2 * DM;
    const float* Wm2 = W_mod2 + (size_t)i * DM * 2 * DM;
    const float* bq = b_q + (size_t)i * DM;
    const float* bk = b_k + (size_t)i * DM;
    const float* bv = b_v + (size_t)i * DM;
    const float* bo = b_o + (size_t)i * DM;
    const float* Wg1 = W_g1 + (size_t)i * DM * DM;  const float* bg1 = b_g1 + (size_t)i * DM;
    const float* Wg2 = W_g2 + (size_t)i * DM * DM;  const float* bg2 = b_g2 + (size_t)i * DM;
    const float* bgg = b_geglu + (size_t)i * 2 * HIDF;
    const float* bff = b_ffout + (size_t)i * DM;
    const _Float16* Wq = Wq_h + (size_t)i * DM * DM;
    const _Float16* Wk = Wk_h + (size_t)i * DM * DM;
    const _Float16* Wv = Wv_h + (size_t)i * DM * DM;
    const _Float16* Wo = Wo_h + (size_t)i * DM * DM;
    const _Float16* Wgg = Wgg_h + (size_t)i * DM * 2 * HIDF;
    const _Float16* Wff = Wff_h + (size_t)i * HIDF * DM;

    // per-frame modulation / gate vectors
    small_mm(stream, silc, Wm1, bm1, mv1c, 6, 2 * DM, DM);
    small_mm(stream, silx, Wm1, bm1, mv1x, 5, 2 * DM, DM);
    small_mm(stream, silc, Wm2, bm2, mv2c, 6, 2 * DM, DM);
    small_mm(stream, silx, Wm2, bm2, mv2x, 5, 2 * DM, DM);
    small_mm(stream, cond,  Wg1, bg1, g1c, 6, DM, DM);
    small_mm(stream, clean, Wg1, bg1, g1x, 5, DM, DM);
    small_mm(stream, cond,  Wg2, bg2, g2c, 6, DM, DM);
    small_mm(stream, clean, Wg2, bg2, g2x, 5, DM, DM);

    // AdaLN 1
    ln_mod_kernel<<<ZR_ROWS, 256, 0, stream>>>(zrn, zrn_h, zr, mv1c, SEQ);
    ln_mod_kernel<<<XA_ROWS, 256, 0, stream>>>(xan, xan_h, xa, mv1x, SEQ);

    // QKV projections (kv concat = [zrn ; xan]) -> f16 outputs
    gemm(stream, zrn_h, Wq, nullptr, qz_h, bq, ZR_ROWS, DM, DM, DM, DM, DM, 1.f, 0);
    gemm(stream, xan_h, Wq, nullptr, qx_h, bq, XA_ROWS, DM, DM, DM, DM, DM, 1.f, 0);
    gemm(stream, zrn_h, Wk, nullptr, kall_h, bk, ZR_ROWS, DM, DM, DM, DM, DM, 1.f, 0);
    gemm(stream, xan_h, Wk, nullptr, kall_h + (size_t)ZR_ROWS * DM, bk, XA_ROWS, DM, DM, DM, DM, DM, 1.f, 0);
    gemm(stream, zrn_h, Wv, nullptr, vall_h, bv, ZR_ROWS, DM, DM, DM, DM, DM, 1.f, 0);
    gemm(stream, xan_h, Wv, nullptr, vall_h + (size_t)ZR_ROWS * DM, bv, XA_ROWS, DM, DM, DM, DM, DM, 1.f, 0);

    const float scale = 0.125f;  // 1/sqrt(64)
    for (int h = 0; h < NH; ++h) {
      size_t ho = (size_t)h * DH;
      // ---- cross attention: q = zr tokens, kv = [zr ; xa]
      gemm(stream, qz_h + ho, kall_h + ho, scores, nullptr, nullptr,
           ZR_ROWS, KV_ROWS, DH, DM, DM, SPAD_C, scale, 1);
      softmax_mask_kernel<<<ZR_ROWS, 256, 0, stream>>>(scores, prob_h, KV_ROWS, SPAD_C, 0);
      gemm(stream, prob_h, vall_h + ho, nullptr, attnz_h + ho, nullptr,
           ZR_ROWS, DH, KV_ROWS, SPAD_C, DM, DM, 1.f, 0);
      // ---- self attention on xa
      gemm(stream, qx_h + ho, kall_h + (size_t)ZR_ROWS * DM + ho, scores, nullptr, nullptr,
           XA_ROWS, XA_ROWS, DH, DM, DM, SPAD_S, scale, 1);
      softmax_mask_kernel<<<XA_ROWS, 256, 0, stream>>>(scores, prob_h, XA_ROWS, SPAD_S, 1);
      gemm(stream, prob_h, vall_h + (size_t)ZR_ROWS * DM + ho, nullptr, attnx_h + ho, nullptr,
           XA_ROWS, DH, XA_ROWS, SPAD_S, DM, DM, 1.f, 0);
    }

    // output projection -> f32
    gemm(stream, attnz_h, Wo, tmpz, nullptr, bo, ZR_ROWS, DM, DM, DM, DM, DM, 1.f, 0);
    gemm(stream, attnx_h, Wo, tmpx, nullptr, bo, XA_ROWS, DM, DM, DM, DM, DM, 1.f, 0);

    // zr = gate1(zrn) + cross ; xa = gate1(xan) + self
    gate_add_kernel<<<ZR_ROWS, 256, 0, stream>>>(zr, zrn, g1c, tmpz, SEQ);
    gate_add_kernel<<<XA_ROWS, 256, 0, stream>>>(xa, xan, g1x, tmpx, SEQ);

    // AdaLN 2
    ln_mod_kernel<<<ZR_ROWS, 256, 0, stream>>>(zrn, zrn_h, zr, mv2c, SEQ);
    ln_mod_kernel<<<XA_ROWS, 256, 0, stream>>>(xan, xan_h, xa, mv2x, SEQ);

    // GEGLU FFN, zr path
    gemm(stream, zrn_h, Wgg, hid, nullptr, bgg, ZR_ROWS, 2 * HIDF, DM, DM, 2 * HIDF, 2 * HIDF, 1.f, 0);
    geglu_act_kernel<<<ZR_ROWS, 256, 0, stream>>>(hact_h, hid);
    gemm(stream, hact_h, Wff, tmpz, nullptr, bff, ZR_ROWS, DM, HIDF, HIDF, DM, DM, 1.f, 0);
    gate_add_kernel<<<ZR_ROWS, 256, 0, stream>>>(zr, tmpz, g2c, nullptr, SEQ);

    // GEGLU FFN, xa path
    gemm(stream, xan_h, Wgg, hid, nullptr, bgg, XA_ROWS, 2 * HIDF, DM, DM, 2 * HIDF, 2 * HIDF, 1.f, 0);
    geglu_act_kernel<<<XA_ROWS, 256, 0, stream>>>(hact_h, hid);
    gemm(stream, hact_h, Wff, tmpx, nullptr, bff, XA_ROWS, DM, HIDF, HIDF, DM, DM, 1.f, 0);
    gate_add_kernel<<<XA_ROWS, 256, 0, stream>>>(xa, tmpx, g2x, nullptr, SEQ);
  }

  unpatch_kernel<<<(6 * 3 * 32 * 32 + 255) / 256, 256, 0, stream>>>(
      zr, W_unpatch, b_unpatch, (float*)d_out);
}
